// EfficientAttention_635655160543
// MI455X (gfx1250) — matrix operand order repaired and checked by reading, not compiled, not run on hardware
//
#include <hip/hip_runtime.h>

// ---------------------------------------------------------------- constants
#define BATCH   8
#define NTOK    4096      // N = 64*64
#define CDIM    512
#define HEADS   8
#define HDIM    64
#define DRED    1024      // (64/2)*(64/2)
#define SCALE_F 0.125f    // 64^-0.5 folded into q projection

typedef __attribute__((ext_vector_type(16))) __bf16 v16bf;
typedef __attribute__((ext_vector_type(8)))  float  v8f;
typedef unsigned int v4u __attribute__((ext_vector_type(4)));
typedef int          v4i __attribute__((ext_vector_type(4)));
typedef int          v8i __attribute__((ext_vector_type(8)));

union Frag { v16bf v; unsigned u[8]; };

#ifndef __has_builtin
#define __has_builtin(x) 0
#endif
#if __has_builtin(__builtin_amdgcn_tensor_load_to_lds) && \
    __has_builtin(__builtin_amdgcn_s_wait_tensorcnt)
#define HAVE_TDM 1
#else
#define HAVE_TDM 0
#endif

__device__ __forceinline__ unsigned short f2bf(float f) {
  union { __bf16 h; unsigned short s; } c;
  c.h = (__bf16)f;                       // hardware cvt, RNE
  return c.s;
}
__device__ __forceinline__ unsigned pack_bf16(float a, float b) {
  return (unsigned)f2bf(a) | ((unsigned)f2bf(b) << 16);
}
// A-matrix (16-bit, 16x32) K index for u32 slot j, lane-half h
__device__ __forceinline__ int a_k(int j, int h) {
  return ((j >= 4) ? 16 : 0) + h * 8 + ((j & 3) << 1);
}
// B-matrix (16-bit, 32x16) K index for u32 slot j, lane-half h
__device__ __forceinline__ int b_k(int j, int h) {
  return ((j >= 4) ? 16 : 0) + h * 8 + ((j & 3) << 1);
}

// ---------------------------------------------------------------- f32 -> bf16
__global__ void __launch_bounds__(256)
cvt_f32_bf16(const float* __restrict__ src, unsigned short* __restrict__ dst,
             int n) {
  int i = (blockIdx.x * 256 + threadIdx.x) * 8;
  if (i + 8 > n) return;
  unsigned* d = (unsigned*)(dst + i);
#pragma unroll
  for (int j = 0; j < 4; ++j) d[j] = pack_bf16(src[i + 2 * j], src[i + 2 * j + 1]);
}

// ---------------------------------------------------------------- GEMM
// OUT[Mr,Nc] = alpha * (A[Mr,K] @ W[Nc,K]^T) + bias.  A,W are bf16.
// GATHER=1: A gathered as SR-conv patches (K order ci*4+ky*2+kx = OIHW flat).
// OUT_BF16: store bf16, else f32.
template <int GATHER, int OUT_BF16>
__global__ void __launch_bounds__(128)
gemm_wmma(const unsigned short* __restrict__ A,
          const unsigned short* __restrict__ W,
          const float* __restrict__ bias, void* __restrict__ outp,
          int Mr, int Nc, int K, float alpha) {
  __shared__ unsigned short As[64][68];   // 136B row stride: dword-aligned, conflict-free
  __shared__ unsigned short Ws[64][68];
  const int tid  = threadIdx.x;
  const int lane = tid & 31, wv = tid >> 5;
  const int half = lane >> 4, l15 = lane & 15;
  const int row0 = blockIdx.x * 64, col0 = blockIdx.y * 64;
  const int wr = (wv >> 1) * 32, wc = (wv & 1) * 32;

  v8f acc[2][2] = {};

  for (int k0 = 0; k0 < K; k0 += 64) {
    if (!GATHER && k0 + 64 < K) {   // prefetch next K tile (global_prefetch_b8)
      __builtin_prefetch(&A[(size_t)(row0 + (tid >> 1)) * K + k0 + 64 + (tid & 1) * 32], 0, 1);
      __builtin_prefetch(&W[(size_t)(col0 + (tid >> 1)) * K + k0 + 64 + (tid & 1) * 32], 0, 1);
    }
    // stage W tile 64x64 bf16 (u32 copies)
#pragma unroll
    for (int j = 0; j < 16; ++j) {
      int ee = j * 128 + tid;
      int r = ee >> 5, ku = ee & 31;
      *(unsigned*)&Ws[r][ku << 1] =
          *(const unsigned*)&W[(size_t)(col0 + r) * K + k0 + (ku << 1)];
    }
    // stage A tile
    if (GATHER) {
#pragma unroll
      for (int j = 0; j < 32; ++j) {
        int ee = j * 128 + tid;
        int r = ee >> 6, kk = ee & 63;
        int kg = k0 + kk;
        int ci = kg >> 2, ky = (kg >> 1) & 1, kx = kg & 1;
        int gr = row0 + r;
        int b = gr >> 10, d = gr & 1023;
        int y = d >> 5, x = d & 31;
        int token = ((y << 1) + ky) * 64 + ((x << 1) + kx);
        As[r][kk] = A[((size_t)(b * NTOK + token)) * CDIM + ci];
      }
    } else {
#pragma unroll
      for (int j = 0; j < 16; ++j) {
        int ee = j * 128 + tid;
        int r = ee >> 5, ku = ee & 31;
        *(unsigned*)&As[r][ku << 1] =
            *(const unsigned*)&A[(size_t)(row0 + r) * K + k0 + (ku << 1)];
      }
    }
    __syncthreads();

#pragma unroll
    for (int ks = 0; ks < 2; ++ks) {
      Frag a[2], b[2];
#pragma unroll
      for (int rt = 0; rt < 2; ++rt) {
        int r = wr + rt * 16 + l15;
#pragma unroll
        for (int j = 0; j < 8; ++j)
          a[rt].u[j] = *(const unsigned*)&As[r][ks * 32 + a_k(j, half)];
      }
#pragma unroll
      for (int ct = 0; ct < 2; ++ct) {
        int cidx = wc + ct * 16 + l15;
#pragma unroll
        for (int j = 0; j < 8; ++j)
          b[ct].u[j] = *(const unsigned*)&Ws[cidx][ks * 32 + b_k(j, half)];
      }
#pragma unroll
      for (int rt = 0; rt < 2; ++rt)
#pragma unroll
        for (int ct = 0; ct < 2; ++ct)
          acc[rt][ct] = __builtin_amdgcn_wmma_f32_16x16x32_bf16(
              false, a[rt].v, false, b[ct].v, (short)0, acc[rt][ct], false, false);
    }
    __syncthreads();
  }

#pragma unroll
  for (int rt = 0; rt < 2; ++rt)
#pragma unroll
    for (int ct = 0; ct < 2; ++ct) {
      int cc = col0 + wc + ct * 16 + l15;
      float bv = bias ? bias[cc] : 0.f;
#pragma unroll
      for (int i = 0; i < 8; ++i) {
        int r = row0 + wr + rt * 16 + i + 8 * half;
        float val = acc[rt][ct][i] * alpha + bv;
        if (OUT_BF16)
          ((unsigned short*)outp)[(size_t)r * Nc + cc] = f2bf(val);
        else
          ((float*)outp)[(size_t)r * Nc + cc] = val;
      }
    }
}

// ---------------------------------------------------------------- LayerNorm
// reads f32, writes bf16 (feeds kv GEMM)
__global__ void __launch_bounds__(128)
layernorm_bf16(const float* __restrict__ xs, const float* __restrict__ g,
               const float* __restrict__ bta, unsigned short* __restrict__ outb,
               int rows) {
  int row = blockIdx.x * 4 + (threadIdx.x >> 5);
  int lane = threadIdx.x & 31;
  if (row >= rows) return;
  const float* p = xs + (size_t)row * CDIM + lane * 16;
  float v[16], s = 0.f, s2 = 0.f;
#pragma unroll
  for (int j = 0; j < 16; ++j) {
    v[j] = p[j];
    s += v[j]; s2 += v[j] * v[j];
  }
#pragma unroll
  for (int m = 16; m; m >>= 1) {
    s  += __shfl_xor(s,  m, 32);
    s2 += __shfl_xor(s2, m, 32);
  }
  float mu = s * (1.f / CDIM);
  float var = s2 * (1.f / CDIM) - mu * mu;
  float rs = rsqrtf(var + 1e-5f);
  const float* gp = g + lane * 16;
  const float* bp = bta + lane * 16;
  unsigned* d = (unsigned*)(outb + (size_t)row * CDIM + lane * 16);
#pragma unroll
  for (int j = 0; j < 8; ++j) {
    float a0 = (v[2 * j]     - mu) * rs * gp[2 * j]     + bp[2 * j];
    float a1 = (v[2 * j + 1] - mu) * rs * gp[2 * j + 1] + bp[2 * j + 1];
    d[j] = pack_bf16(a0, a1);
  }
}

// ---------------------------------------------------------------- attention
// Flash attention, all-bf16 inputs; q pre-scaled by 0.125 in its GEMM.
// K chunk staged by Tensor Data Mover (64x64 bf16 tile, LDS-padded rows);
// V staged transposed by VALU path.
__global__ void __launch_bounds__(128)
attn_wmma(const unsigned short* __restrict__ q,
          const unsigned short* __restrict__ kv,
          unsigned short* __restrict__ out) {
  __shared__ unsigned short Ks[64][68];       // [d][c], 2-dword pad per 32-dword row
  __shared__ unsigned short VsT[64][68];      // [c][d] (transposed)
  __shared__ unsigned short Pw[4][16][68];    // per-wave P relayout
  const int tid = threadIdx.x, lane = tid & 31, wv = tid >> 5;
  const int half = lane >> 4, l15 = lane & 15;
  const int b = blockIdx.z, m = blockIdx.y;
  const int n0 = blockIdx.x * 64 + wv * 16;

  // q A-fragments (K = 64 -> 2 fragments), straight u32 loads of bf16 pairs
  Frag qa[2];
  {
    const unsigned short* qp = q + ((size_t)(b * NTOK + n0 + l15)) * CDIM + m * HDIM;
#pragma unroll
    for (int ks = 0; ks < 2; ++ks)
#pragma unroll
      for (int j = 0; j < 8; ++j)
        qa[ks].u[j] = *(const unsigned*)&qp[ks * 32 + a_k(j, half)];
  }

  v8f acc[4] = {};
  float mrun[8], lrun[8];
#pragma unroll
  for (int i = 0; i < 8; ++i) { mrun[i] = -1e30f; lrun[i] = 0.f; }

  const unsigned short* kvb = kv + (size_t)b * DRED * 1024;  // rows D, 1024 bf16 cols

  for (int d0 = 0; d0 < DRED; d0 += 64) {
    const unsigned short* src0 = kvb + (size_t)d0 * 1024 + m * HDIM;
#if HAVE_TDM
    if (wv == 0) {
      // Tensor DMA: 64x64 2-byte tile, row stride 1024 elems, pad 2 dwords / 32 dwords
      unsigned long long ga = (unsigned long long)(uintptr_t)src0;
      v4u g0;
      g0.x = 1u;                                    // count=1, user descriptor
      g0.y = (unsigned)(uintptr_t)&Ks[0][0];        // LDS offset (low 32 bits of flat)
      g0.z = (unsigned)ga;
      g0.w = (unsigned)(ga >> 32) | (2u << 30);     // addr[56:32] | type=2
      v8i g1;
      g1.s0 = (int)((1u << 16) | (1u << 20) | (4u << 22) | (1u << 25));
      g1.s1 = (int)(64u << 16);                     // tensor_dim0 = 64
      g1.s2 = (int)(64u << 16);                     // tensor_dim1 = 64
      g1.s3 = (int)(64u << 16);                     // tile_dim0 = 64
      g1.s4 = 64;                                   // tile_dim1 = 64
      g1.s5 = 1024;                                 // tensor_dim0_stride
      g1.s6 = 0; g1.s7 = 0;
      v4i gz = {};
#if __clang_major__ >= 23
      v8i gz8 = {};
      __builtin_amdgcn_tensor_load_to_lds(g0, g1, gz, gz, gz8, 0);
#else
      __builtin_amdgcn_tensor_load_to_lds(g0, g1, gz, gz, 0);
#endif
    }
#else
    // fallback: stage K with u32 copies
#pragma unroll
    for (int j = 0; j < 16; ++j) {
      int ee = j * 128 + tid;
      int r = ee >> 5, ku = ee & 31;
      *(unsigned*)&Ks[r][ku << 1] =
          *(const unsigned*)&src0[(size_t)r * 1024 + (ku << 1)];
    }
#endif
    // stage V transposed (bf16 pair split)
#pragma unroll
    for (int j = 0; j < 16; ++j) {
      int ee = j * 128 + tid;
      int r = ee >> 5, ku = ee & 31;
      unsigned vw = *(const unsigned*)&src0[(size_t)r * 1024 + CDIM + (ku << 1)];
      VsT[(ku << 1)][r]     = (unsigned short)(vw & 0xffffu);
      VsT[(ku << 1) + 1][r] = (unsigned short)(vw >> 16);
    }
#if HAVE_TDM
    if (wv == 0) __builtin_amdgcn_s_wait_tensorcnt(0);
#endif
    __syncthreads();

    // S = q @ K^T  (16 x 64 per wave), scale pre-folded into q
    v8f s[4] = {};
#pragma unroll
    for (int ct = 0; ct < 4; ++ct) {
      int dd = ct * 16 + l15;
#pragma unroll
      for (int ks = 0; ks < 2; ++ks) {
        Frag bf;
#pragma unroll
        for (int j = 0; j < 8; ++j)
          bf.u[j] = *(const unsigned*)&Ks[dd][ks * 32 + b_k(j, half)];
        s[ct] = __builtin_amdgcn_wmma_f32_16x16x32_bf16(
            false, qa[ks].v, false, bf.v, (short)0, s[ct], false, false);
      }
    }

    // row-max + online-softmax update
    float rm[8];
#pragma unroll
    for (int i = 0; i < 8; ++i) {
      float mx = fmaxf(fmaxf(s[0][i], s[1][i]), fmaxf(s[2][i], s[3][i]));
#pragma unroll
      for (int msk = 1; msk < 16; msk <<= 1)
        mx = fmaxf(mx, __shfl_xor(mx, msk, 32));
      rm[i] = mx;
    }
#pragma unroll
    for (int i = 0; i < 8; ++i) {
      float mn = fmaxf(mrun[i], rm[i]);
      float sc = __expf(mrun[i] - mn);
      mrun[i] = mn;
      lrun[i] *= sc;
#pragma unroll
      for (int ct = 0; ct < 4; ++ct) acc[ct][i] *= sc;
    }

    // P = exp(S - m), row sums, stash P (bf16) for A-fragment relayout
#pragma unroll
    for (int i = 0; i < 8; ++i) {
      int prow = i + 8 * half;
      float rs = 0.f;
#pragma unroll
      for (int ct = 0; ct < 4; ++ct) {
        float p = __expf(s[ct][i] - mrun[i]);
        rs += p;
        Pw[wv][prow][ct * 16 + l15] = f2bf(p);
      }
#pragma unroll
      for (int msk = 1; msk < 16; msk <<= 1) rs += __shfl_xor(rs, msk, 32);
      lrun[i] += rs;
    }

    // O += P @ V  (K = d-chunk = 64)
#pragma unroll
    for (int ks = 0; ks < 2; ++ks) {
      Frag pa;
#pragma unroll
      for (int j = 0; j < 8; ++j)
        pa.u[j] = *(const unsigned*)&Pw[wv][l15][ks * 32 + a_k(j, half)];
#pragma unroll
      for (int ct = 0; ct < 4; ++ct) {
        Frag vb;
        int ccc = ct * 16 + l15;
#pragma unroll
        for (int j = 0; j < 8; ++j)
          vb.u[j] = *(const unsigned*)&VsT[ccc][ks * 32 + b_k(j, half)];
        acc[ct] = __builtin_amdgcn_wmma_f32_16x16x32_bf16(
            false, pa.v, false, vb.v, (short)0, acc[ct], false, false);
      }
    }
    __syncthreads();
  }

  // normalize + write O (bf16) to [B, N, C] at column m*64+cc
#pragma unroll
  for (int ct = 0; ct < 4; ++ct) {
    int cc = ct * 16 + l15;
#pragma unroll
    for (int i = 0; i < 8; ++i) {
      int n = n0 + i + 8 * half;
      out[((size_t)(b * NTOK + n)) * CDIM + m * HDIM + cc] = f2bf(acc[ct][i] / lrun[i]);
    }
  }
}

// ---------------------------------------------------------------- launcher
extern "C" void kernel_launch(void* const* d_in, const int* in_sizes, int n_in,
                              void* d_out, int out_size, void* d_ws, size_t ws_size,
                              hipStream_t stream) {
  const float* x      = (const float*)d_in[0];
  const float* q_w    = (const float*)d_in[3];
  const float* kv_w   = (const float*)d_in[4];
  const float* proj_w = (const float*)d_in[5];
  const float* proj_b = (const float*)d_in[6];
  const float* sr_w   = (const float*)d_in[7];   // [512][2048] OIHW flat
  const float* sr_b   = (const float*)d_in[8];
  const float* norm_g = (const float*)d_in[9];
  const float* norm_b = (const float*)d_in[10];
  float* outp = (float*)d_out;

  // workspace partition
  unsigned short* xb  = (unsigned short*)d_ws;         // [32768,512]
  unsigned short* wsr = xb  + (size_t)32768 * 512;     // [512,2048]
  unsigned short* wq  = wsr + (size_t)512 * 2048;      // [512,512]
  unsigned short* wkv = wq  + (size_t)512 * 512;       // [1024,512]
  unsigned short* wpj = wkv + (size_t)1024 * 512;      // [512,512]
  unsigned short* qb  = wpj + (size_t)512 * 512;       // [32768,512]
  unsigned short* kvb = qb  + (size_t)32768 * 512;     // [8192,1024]
  unsigned short* xsb = kvb + (size_t)8192 * 1024;     // [8192,512]
  unsigned short* aob = xsb + (size_t)8192 * 512;      // [32768,512]
  float*          xs  = (float*)(aob + (size_t)32768 * 512);  // [8192,512] f32

  dim3 blk(128);

  // 0) one-shot f32 -> bf16 conversions
  cvt_f32_bf16<<<dim3(32768 * 512 / 2048), dim3(256), 0, stream>>>(x, xb, 32768 * 512);
  cvt_f32_bf16<<<dim3(512 * 2048 / 2048), dim3(256), 0, stream>>>(sr_w, wsr, 512 * 2048);
  cvt_f32_bf16<<<dim3(512 * 512 / 2048), dim3(256), 0, stream>>>(q_w, wq, 512 * 512);
  cvt_f32_bf16<<<dim3(1024 * 512 / 2048), dim3(256), 0, stream>>>(kv_w, wkv, 1024 * 512);
  cvt_f32_bf16<<<dim3(512 * 512 / 2048), dim3(256), 0, stream>>>(proj_w, wpj, 512 * 512);

  // 1) SR conv as implicit GEMM (f32 out for LayerNorm)
  gemm_wmma<1, 0><<<dim3(8192 / 64, 512 / 64), blk, 0, stream>>>(
      xb, wsr, sr_b, xs, 8192, 512, 2048, 1.f);

  // 2) LayerNorm -> bf16
  layernorm_bf16<<<dim3(8192 / 4), blk, 0, stream>>>(xs, norm_g, norm_b, xsb, 8192);

  // 3) q = (x @ q_w^T) * SCALE  (scale folded here)
  gemm_wmma<0, 1><<<dim3(32768 / 64, 512 / 64), blk, 0, stream>>>(
      xb, wq, nullptr, qb, 32768, 512, 512, SCALE_F);

  // 4) kv = xs_ln @ kv_w^T
  gemm_wmma<0, 1><<<dim3(8192 / 64, 1024 / 64), blk, 0, stream>>>(
      xsb, wkv, nullptr, kvb, 8192, 1024, 512, 1.f);

  // 5) flash attention (bf16 in/out)
  attn_wmma<<<dim3(4096 / 64, HEADS, BATCH), blk, 0, stream>>>(qb, kvb, aob);

  // 6) out = ao @ proj_w^T + proj_b (f32 out)
  gemm_wmma<0, 0><<<dim3(32768 / 64, 512 / 64), blk, 0, stream>>>(
      aob, wpj, proj_b, outp, 32768, 512, 512, 1.f);
}
